// MoeAgScatterOp_86646670229700
// MI455X (gfx1250) — matrix operand order repaired and checked by reading, not compiled, not run on hardware
//
#include <hip/hip_runtime.h>

// ---------------------------------------------------------------------------
// MoE AG-scatter grouped GEMM for MI455X (gfx1250, wave32, WMMA + TDM).
// out[s, :] = x[token(s), :] . W[expert(s), :, :]^T
// bf16 WMMA (v_wmma_f32_16x16x32_bf16), fp32 accum.  B (weights) tiles are
// DMA'd to LDS by the Tensor Data Mover (tensor_load_to_lds, TENSORcnt);
// A (token-gathered activations) staged via global loads + ds_store.
// Double-buffered LDS, BK=64 -> 16 WMMAs per barrier.
// ---------------------------------------------------------------------------

#define NTOK  8192
#define TOPK  2
#define NEXP  8
#define KD    1024
#define ND    1024
#define MOUT  (NTOK * TOPK)          // 16384 dispatched rows

#define BM 128
#define BN 128
#define BK 64
#define NKITER (KD / BK)             // 16 stage iterations
#define LDSS 72                      // BK + 8 halves pad (144B row, 16B aligned)
#define NT (ND / BN)                 // 8 N-tiles
#define MAX_MT ((MOUT / BM) + NEXP)  // 136 worst-case expert-segment M tiles
#define MAX_TILES 160

typedef __attribute__((ext_vector_type(16))) __bf16 v16bf;
typedef __attribute__((ext_vector_type(8)))  float  v8f;
typedef unsigned short u16;
typedef unsigned int   u32;
typedef __attribute__((ext_vector_type(4))) unsigned int uint32x4;
typedef __attribute__((ext_vector_type(4))) int          int32x4;
typedef __attribute__((ext_vector_type(8))) int          int32x8;

union Frag16 { uint4 q[2]; v16bf v; };   // 32B: one 16-bit WMMA A/B operand

#if defined(__has_builtin)
#  if __has_builtin(__builtin_amdgcn_tensor_load_to_lds)
#    define HAVE_TDM 1
#  endif
#endif
#if defined(__has_include)
#  if __has_include(<hip/amd_detail/amd_gfx1250_TDM.h>)
#    define TDM_6ARG 1   // amdgpu-toolchain (therock-10.0) 6-arg builtin
#  endif
#endif

__device__ __forceinline__ u16 f32_to_bf16(float f) {
    u32 u = __float_as_uint(f);
    u32 r = u + 0x7FFFu + ((u >> 16) & 1u);   // round-to-nearest-even
    return (u16)(r >> 16);
}

#ifdef HAVE_TDM
// DMA one 128(N) x 64(K) bf16 tile of W[e] (row stride KD halves) into LDS
// with 4-dword padding after each 32-dword row (-> 72-half LDS row stride).
// D# layout per CDNA5 ISA.  Issued by one wave per WG.
__device__ __forceinline__ void tdm_load_b_tile(const u16* gsrc, u16* lds_dst)
{
    unsigned long long ga = (unsigned long long)(uintptr_t)gsrc;
    unsigned la = (unsigned)(uintptr_t)lds_dst;   // low 32 bits == LDS offset

    uint32x4 g0 = { 1u,                                    // count=1, user D#
                    la,                                    // lds_addr
                    (unsigned)ga,                          // global_addr[31:0]
                    (((unsigned)(ga >> 32)) & 0x01FFFFFFu) // global_addr[56:32]
                        | (2u << 30) };                    // type=2 (image)
    int32x8 g1 = {
        (int)((1u << 16)        // data_size = 2 bytes
            | (1u << 20)        // pad_enable
            | (4u << 22)        // pad_interval: 32 dwords (= tile_dim0 row)
            | (3u << 25)),      // pad_amount: 4 dwords (8 halves)
        (int)((u32)KD << 16),   // tensor_dim0[15:0] (K extent)
        (int)((u32)ND << 16),   // tensor_dim0[31:16]=0 | tensor_dim1[15:0]
        (int)((u32)BK << 16),   // tensor_dim1[31:16]=0 | tile_dim0=64
        (int)BN,                // tile_dim1=128 | tile_dim2=0
        (int)KD,                // tensor_dim0_stride[31:0] = K
        0, 0 };
    int32x4 z4 = { 0, 0, 0, 0 };
#ifdef TDM_6ARG
    int32x8 z8 = { 0, 0, 0, 0, 0, 0, 0, 0 };
    __builtin_amdgcn_tensor_load_to_lds(g0, g1, z4, z4, z8, 0);
#else
    __builtin_amdgcn_tensor_load_to_lds(g0, g1, z4, z4, 0);
#endif
}
#endif // HAVE_TDM

// ---------------------------------------------------------------------------
// Kernel 1: x (f32 [NTOK,KD]) and W (f32 [NEXP,ND,KD]) -> bf16 in workspace.
// ---------------------------------------------------------------------------
__global__ __launch_bounds__(256)
void convert_bf16_kernel(const float* __restrict__ x, const float* __restrict__ w,
                         u16* __restrict__ xb, u16* __restrict__ wb)
{
    const long XN = (long)NTOK * KD;
    const long WN = (long)NEXP * ND * KD;
    long base = ((long)blockIdx.x * blockDim.x + threadIdx.x) * 4;
    if (base < XN) {
        float4 f = *(const float4*)(x + base);
        ushort4 o;
        o.x = f32_to_bf16(f.x); o.y = f32_to_bf16(f.y);
        o.z = f32_to_bf16(f.z); o.w = f32_to_bf16(f.w);
        *(ushort4*)(xb + base) = o;
    } else if (base < XN + WN) {
        long b2 = base - XN;
        float4 f = *(const float4*)(w + b2);
        ushort4 o;
        o.x = f32_to_bf16(f.x); o.y = f32_to_bf16(f.y);
        o.z = f32_to_bf16(f.z); o.w = f32_to_bf16(f.w);
        *(ushort4*)(wb + b2) = o;
    }
}

// ---------------------------------------------------------------------------
// Kernel 2: row_token[scatter_index[i]] = i / TOPK  (invert dispatch perm).
// ---------------------------------------------------------------------------
__global__ __launch_bounds__(256)
void build_row_token_kernel(const int* __restrict__ scatter_index,
                            int* __restrict__ row_token)
{
    int i = blockIdx.x * blockDim.x + threadIdx.x;
    if (i < MOUT) row_token[scatter_index[i]] = i / TOPK;
}

// ---------------------------------------------------------------------------
// Kernel 3: per-expert M-tile table from splits (E=8, trivial on one thread).
// ---------------------------------------------------------------------------
__global__ void build_tiles_kernel(const int* __restrict__ splits,
                                   int* __restrict__ meta,
                                   int4* __restrict__ tiles)
{
    if (threadIdx.x == 0 && blockIdx.x == 0) {
        int off = 0, t = 0;
        for (int e = 0; e < NEXP; ++e) {
            int cnt = splits[e];
            for (int m = 0; m < cnt; m += BM) {
                int rows = cnt - m; if (rows > BM) rows = BM;
                tiles[t] = make_int4(off + m, rows, e, 0);
                ++t;
            }
            off += cnt;
        }
        meta[0] = t;
    }
}

// ---------------------------------------------------------------------------
// Kernel 4: grouped GEMM.  Block = 256 threads (8 wave32s), tile 128x128,
// BK=64, double-buffered LDS.  Wave (wm,wn) owns 32x64 = 2x4 WMMA frags.
//
// 16-bit A 16x32 lane layout: lane L row m=L&15; halves j=0..7 -> K=(L>>4)*8+j,
// j=8..15 -> K=(L>>4)*8+8+j  => two 16B runs at kbase, kbase+16.
// 16-bit B 32x16: lane L col n=L&15; same K order as A (runs at kbase, kbase+16).
// f32 C/D 16x16: lane L col n=L&15; VGPR v -> row m=v+(L>>4)*8.
// ---------------------------------------------------------------------------
__global__ __launch_bounds__(256)
void moe_gemm_kernel(const u16* __restrict__ xb, const u16* __restrict__ wb,
                     const int* __restrict__ row_token,
                     const int* __restrict__ meta,
                     const int4* __restrict__ tiles,
                     float* __restrict__ out)
{
    const int ntiles = meta[0];
    const int mt = blockIdx.y;
    if (mt >= ntiles) return;

    const int4 td = tiles[mt];
    const int row_start = td.x;
    const int rows      = td.y;
    const int expert    = td.z;
    const int nt        = blockIdx.x;

    __shared__ u16 As[2][BM * LDSS];
    __shared__ u16 Bs[2][BN * LDSS];

    const int tid  = threadIdx.x;
    const int lane = tid & 31;
    const int wid  = tid >> 5;
    const int wm   = wid & 3;      // 4 waves along M
    const int wn   = wid >> 2;     // 2 waves along N

    // --- A staging map: thread t loads 32 halves of row (t>>1), cols (t&1)*32
    const int lrow = tid >> 1;
    const int lcol = (tid & 1) * 32;
    const int arow = (lrow < rows) ? lrow : 0;        // pad rows -> safe token
    const u16* a_g = xb + (long)row_token[row_start + arow] * KD + lcol;
    u16* a_l = &As[0][0] + lrow * LDSS + lcol;        // (+ buffer offset)

    // --- B tile source (dense rows of W[expert]) ---
    const u16* b_tile0 = wb + (long)expert * ND * KD + (long)(nt * BN) * KD;
#ifndef HAVE_TDM
    const u16* b_g = b_tile0 + (long)lrow * KD + lcol;
    u16* b_l = &Bs[0][0] + lrow * LDSS + lcol;
#endif

    // --- fragment read bases ---
    const int m16 = wm * 32 + (lane & 15);
    const int kbA = (lane >> 4) * 8;
    const int n16 = wn * 64 + (lane & 15);
    const int kbB = (lane >> 4) * 8;

    v8f acc[2][4] = {};

    // ---- prologue: stage buffer 0 (kb = 0) ----
    {
#pragma unroll
        for (int j = 0; j < 4; ++j) {
            uint4 v = ((const uint4*)a_g)[j];
            *(uint4*)(a_l + j * 8) = v;
        }
#ifdef HAVE_TDM
        if (wid == 0) {
            tdm_load_b_tile(b_tile0, &Bs[0][0]);
            __builtin_amdgcn_s_wait_tensorcnt(0);
        }
#else
#pragma unroll
        for (int j = 0; j < 4; ++j) {
            uint4 v = ((const uint4*)b_g)[j];
            *(uint4*)(b_l + j * 8) = v;
        }
#endif
        __syncthreads();
    }

    for (int it = 0; it < NKITER; ++it) {
        const int cur = it & 1;
        const int nxt = cur ^ 1;
        const bool have_next = (it + 1) < NKITER;
        const int kb_next = (it + 1) * BK;

        // ---- kick off next tile's data movement before computing ----
        uint4 av[4];
        if (have_next) {
#pragma unroll
            for (int j = 0; j < 4; ++j)
                av[j] = ((const uint4*)(a_g + kb_next))[j];
#ifdef HAVE_TDM
            if (wid == 0)
                tdm_load_b_tile(b_tile0 + kb_next, &Bs[nxt][0]);   // async; wait below
#endif
        }
#ifndef HAVE_TDM
        uint4 bv[4];
        if (have_next) {
#pragma unroll
            for (int j = 0; j < 4; ++j)
                bv[j] = ((const uint4*)(b_g + kb_next))[j];
        }
#endif

        // ---- compute on buffer `cur`: 2 k-substeps x (2x4) WMMAs ----
#pragma unroll
        for (int ks = 0; ks < BK; ks += 32) {
            Frag16 a[2], b[4];
#pragma unroll
            for (int r = 0; r < 2; ++r) {
                const u16* p = &As[cur][(m16 + r * 16) * LDSS + ks + kbA];
                a[r].q[0] = *(const uint4*)(p);
                a[r].q[1] = *(const uint4*)(p + 16);
            }
#pragma unroll
            for (int c = 0; c < 4; ++c) {
                const u16* p = &Bs[cur][(n16 + c * 16) * LDSS + ks + kbB];
                b[c].q[0] = *(const uint4*)(p);
                b[c].q[1] = *(const uint4*)(p + 16);
            }
#pragma unroll
            for (int r = 0; r < 2; ++r)
#pragma unroll
                for (int c = 0; c < 4; ++c)
                    acc[r][c] = __builtin_amdgcn_wmma_f32_16x16x32_bf16(
                        false, a[r].v, false, b[c].v,
                        (short)0, acc[r][c], false, false);
        }

        // ---- publish buffer `nxt` ----
        if (have_next) {
            u16* al = &As[nxt][0] + lrow * LDSS + lcol;
#pragma unroll
            for (int j = 0; j < 4; ++j)
                *(uint4*)(al + j * 8) = av[j];
#ifndef HAVE_TDM
            u16* bl = &Bs[nxt][0] + lrow * LDSS + lcol;
#pragma unroll
            for (int j = 0; j < 4; ++j)
                *(uint4*)(bl + j * 8) = bv[j];
#else
            if (wid == 0)
                __builtin_amdgcn_s_wait_tensorcnt(0);   // TDM done before barrier
#endif
            __syncthreads();
        }
    }

    // ---- writeback (guard rows past the expert segment) ----
    const int nbase = nt * BN + wn * 64;
    const int mhalf = (lane >> 4) * 8;
    const int ncol0 = lane & 15;
#pragma unroll
    for (int r = 0; r < 2; ++r) {
#pragma unroll
        for (int v = 0; v < 8; ++v) {
            const int mloc = wm * 32 + r * 16 + mhalf + v;
            if (mloc < rows) {
                float* orow = out + (long)(row_start + mloc) * ND;
#pragma unroll
                for (int c = 0; c < 4; ++c)
                    orow[nbase + c * 16 + ncol0] = acc[r][c][v];
            }
        }
    }
}

// ---------------------------------------------------------------------------
// Launch wrapper.  Workspace: x bf16 (16MB) | W bf16 (16MB) | row_token | meta
// | tiles.  ~33.6 MB total.
// ---------------------------------------------------------------------------
extern "C" void kernel_launch(void* const* d_in, const int* in_sizes, int n_in,
                              void* d_out, int out_size, void* d_ws, size_t ws_size,
                              hipStream_t stream)
{
    const float* x       = (const float*)d_in[0];
    const float* w       = (const float*)d_in[1];
    const int*   scatter = (const int*)d_in[2];
    const int*   splits  = (const int*)d_in[3];
    float*       out     = (float*)d_out;

    char* ws = (char*)d_ws;
    const size_t XB_OFF = 0;
    const size_t WB_OFF = XB_OFF + (size_t)NTOK * KD * 2;
    const size_t RT_OFF = WB_OFF + (size_t)NEXP * ND * KD * 2;
    const size_t MT_OFF = RT_OFF + (size_t)MOUT * 4;
    const size_t TT_OFF = MT_OFF + 16;

    u16*  xb        = (u16*)(ws + XB_OFF);
    u16*  wb        = (u16*)(ws + WB_OFF);
    int*  row_token = (int*)(ws + RT_OFF);
    int*  meta      = (int*)(ws + MT_OFF);
    int4* tiles     = (int4*)(ws + TT_OFF);

    {
        long total = (long)NTOK * KD + (long)NEXP * ND * KD;  // 16,777,216
        int blocks = (int)(total / 4 / 256);                  // 16384
        convert_bf16_kernel<<<blocks, 256, 0, stream>>>(x, w, xb, wb);
    }
    build_row_token_kernel<<<MOUT / 256, 256, 0, stream>>>(scatter, row_token);
    build_tiles_kernel<<<1, 64, 0, stream>>>(splits, meta, tiles);
    {
        dim3 grid(NT, MAX_MT);
        moe_gemm_kernel<<<grid, 256, 0, stream>>>(xb, wb, row_token, meta, tiles, out);
    }
}
